// DeformAttention_8830452761278
// MI455X (gfx1250) — compile-verified, matrix operand order checked by reading, not run on hardware
//
#include <hip/hip_runtime.h>
#include <cstdint>
#include <cstddef>

// ---------------- types ----------------
typedef __attribute__((ext_vector_type(16))) __bf16 v16bf;
typedef __attribute__((ext_vector_type(8)))  float  v8f;
typedef __attribute__((ext_vector_type(8)))  __bf16 v8bf;
typedef __attribute__((address_space(3)))    __bf16 lds_bf16_t;

union BF16x16 { v16bf v; uint4 q[2]; };
union BF16x8  { v8bf  v; uint4 q;    };

static __device__ __forceinline__ int imin(int a, int b) { return a < b ? a : b; }
static __device__ __forceinline__ int imax(int a, int b) { return a > b ? a : b; }

// ---------------- pack: NCHW fp32 (src||tgt) -> NHWC bf16 [2,192,192,128] ----------------
__global__ __launch_bounds__(256) void pack_input_nhwc(
    const float* __restrict__ src, const float* __restrict__ tgt,
    __bf16* __restrict__ out, int total)
{
    int i = blockIdx.x * 256 + threadIdx.x;
    if (i >= total) return;
    int c   = i & 127;
    int pix = i >> 7;
    int x   = pix % 192;
    int y   = (pix / 192) % 192;
    int b   = pix / (192 * 192);
    float v = (c < 64)
        ? src[(((size_t)b * 64 + c)        * 192 + y) * 192 + x]
        : tgt[(((size_t)b * 64 + (c - 64)) * 192 + y) * 192 + x];
    out[i] = (__bf16)v;
}

// ---------------- pack: OIHW fp32 weights -> per-WMMA-fragment bf16 ----------------
// Fragment = 16 couts x 32 K (K enumerated as tap kk in [0,9) x cin-block of 32).
// Storage: frag-major, then lane (0..31), then 16 bf16 per lane as two runs of 8:
//   elem = run*8 + wi  ->  Krel = run*16 + (lane>>4)*8 + wi
__global__ __launch_bounds__(256) void pack_weights_frag(
    const float* __restrict__ W, __bf16* __restrict__ out,
    int Cout, int Cin, int total)
{
    int i = blockIdx.x * 256 + threadIdx.x;
    if (i >= total) return;
    int elem = i & 15;
    int lane = (i >> 4) & 31;
    int frag = i >> 9;
    int CinBlk = Cin >> 5;
    int Ksteps = 9 * CinBlk;
    int s     = frag % Ksteps;
    int mtile = frag / Ksteps;
    int kk = s / CinBlk;
    int cb = s % CinBlk;
    int m    = lane & 15;
    int half = lane >> 4;
    int run  = elem >> 3;
    int wi   = elem & 7;
    int Krel = run * 16 + half * 8 + wi;
    int cin  = cb * 32 + Krel;
    int cout = mtile * 16 + m;
    float v = (cout < Cout) ? W[((size_t)cout * Cin + cin) * 9 + kk] : 0.0f;
    out[i] = (__bf16)v;
}

// =====================================================================================
// Main implicit-GEMM 3x3 conv via WMMA with workgroup-cooperative LDS staging of the
// activation (B) tile through the CDNA5 async global->LDS path. All 8 waves of a WG
// share one 64-pixel N-tile; each wave owns 2 M-tiles (32 couts) x 4 N-subtiles
// = 8 WMMAs per K-step.
// IN_MODE: 0 = input is [G, P, P, Cin] bf16 channels-last, per-image zero pad
//          1 = patch-gather from [2,192,192,128] (patch 24, stride 12, 15x15 grid)
// Output: ReLU -> bf16 channels-last [Npix, Cout].
// =====================================================================================
template<int IN_MODE>
__global__ __launch_bounds__(256) void conv3x3_wmma_lds(
    const __bf16* __restrict__ in,
    const __bf16* __restrict__ wp,
    const float*  __restrict__ bias,
    __bf16*       __restrict__ out,
    int P, int Cin, int Cout, int Mtiles, int MB)
{
    // double-buffered B tile: 2 x (64 px x 32 cin) bf16 = 8 KB
    __shared__ alignas(16) __bf16 lds[2 * 2048];

    const int tid  = threadIdx.x;
    const int lane = tid & 31;
    // force wave id into an SGPR so all M-tile predicates are provably wave-uniform
    // (WMMA requires EXEC all-ones; we must get s_cbranch, not saveexec, guards)
    const int wid  = __builtin_amdgcn_readfirstlane(tid >> 5);
    const int  mgroup = blockIdx.x % MB;
    const long ntile  = blockIdx.x / MB;

    const int half = lane >> 4;
    const int nl   = lane & 15;
    const int CinBlk = Cin >> 5;
    const int Ksteps = 9 * CinBlk;
    const int PP = P * P;
    const int Wsrc = (IN_MODE == 1) ? 192 : P;    // source row pitch (pixels)
    const int Cst  = (IN_MODE == 1) ? 128 : Cin;  // source channel count

    // ---- staging geometry: this thread owns one 16B piece of the 4KB B tile ----
    const int ss    = tid >> 6;         // sub-tile 0..3
    const int sl    = (tid >> 1) & 31;  // fragment lane
    const int srun  = tid & 1;          // K-run 0/1
    const int shalf = sl >> 4;
    const int snl   = sl & 15;
    long spix = ntile * 64 + ss * 16 + snl;
    int sg = (int)(spix / PP);
    int sw = (int)(spix % PP);
    int spy = sw / P, spx = sw % P;
    const __bf16* sbase;
    if (IN_MODE == 1) {
        int b = sg / 225; int r = sg % 225;
        int row = (r / 15) * 12 + spy;
        int col = (r % 15) * 12 + spx;
        sbase = in + (((size_t)b * 192 + row) * 192 + col) * 128;
    } else {
        sbase = in + ((size_t)sg * PP + (size_t)spy * P + spx) * (size_t)Cin;
    }
    const int sdst = ss * 512 + sl * 16 + srun * 8;   // element offset within buffer

    // ---- wave compute geometry: 2 consecutive M-tiles (all wave-uniform scalars) ----
    const int  mt0  = mgroup * 16 + wid * 2;
    const int  mt1  = mt0 + 1;
    const bool has0 = (mt0 < Mtiles);
    const bool has1 = (mt1 < Mtiles);

    v8f acc[2][4] = {};

    const __bf16* w0 = wp + ((size_t)mt0 * Ksteps) * 512 + lane * 16;
    const __bf16* w1 = wp + ((size_t)mt1 * Ksteps) * 512 + lane * 16;

    int iter = 0;
    for (int kk = 0; kk < 9; ++kk) {
        const int dy = kk / 3 - 1;
        const int dx = kk % 3 - 1;
        const bool svalid = (spy + dy >= 0) && (spy + dy < P) &&
                            (spx + dx >= 0) && (spx + dx < P);
        const __bf16* srow = sbase + ((long)dy * Wsrc + dx) * Cst;
        for (int cb = 0; cb < CinBlk; ++cb, ++iter) {
            // ---- stage 64px x 32cin block into LDS ----
            // valid lanes: async global->LDS copy (ASYNCcnt); padded lanes: zero via DS
            const int bufo = (iter & 1) ? 2048 : 0;
            __bf16* dstp = &lds[bufo + sdst];
            if (svalid) {
                const __bf16* gp = srow + cb * 32 + srun * 16 + shalf * 8;
                unsigned ldsoff = (unsigned)(uintptr_t)(lds_bf16_t*)dstp;
                unsigned long long gaddr = (unsigned long long)(uintptr_t)gp;
                asm volatile("global_load_async_to_lds_b128 %0, %1, off"
                             :: "v"(ldsoff), "v"(gaddr)
                             : "memory");
            } else {
                uint4 z = {0, 0, 0, 0};
                *(uint4*)dstp = z;
            }
            asm volatile("s_wait_asynccnt 0" ::: "memory");
            __syncthreads();

            if (has0) {
                // A fragments: 32 contiguous bytes per lane per M-tile
                BF16x16 a0, a1;
                {
                    const uint4* ap = (const uint4*)(w0 + (size_t)(kk * CinBlk + cb) * 512);
                    a0.q[0] = ap[0]; a0.q[1] = ap[1];
                }
                if (has1) {
                    const uint4* ap = (const uint4*)(w1 + (size_t)(kk * CinBlk + cb) * 512);
                    a1.q[0] = ap[0]; a1.q[1] = ap[1];
                }
#pragma unroll
                for (int s = 0; s < 4; ++s) {
                    BF16x16 b;
                    const __bf16* bp = &lds[bufo + s * 512 + lane * 16];
                    b.q[0] = *(const uint4*)bp;
                    b.q[1] = *(const uint4*)(bp + 8);
                    acc[0][s] = __builtin_amdgcn_wmma_f32_16x16x32_bf16(
                        false, a0.v, false, b.v, (short)0, acc[0][s], false, false);
                    if (has1)
                        acc[1][s] = __builtin_amdgcn_wmma_f32_16x16x32_bf16(
                            false, a1.v, false, b.v, (short)0, acc[1][s], false, false);
                }
            }
        }
    }

    // ---- epilogue: bias + ReLU + cvt to bf16, one b128 store per sub-tile per mtile ----
#pragma unroll
    for (int mi = 0; mi < 2; ++mi) {
        if ((mi == 0 && !has0) || (mi == 1 && !has1)) continue;
        int mt = (mi == 0) ? mt0 : mt1;
        const float4* bp4 = (const float4*)(bias + mt * 16 + half * 8);
        float4 bl = bp4[0], bh = bp4[1];
        float bv[8] = { bl.x, bl.y, bl.z, bl.w, bh.x, bh.y, bh.z, bh.w };
#pragma unroll
        for (int s = 0; s < 4; ++s) {
            BF16x8 o;
#pragma unroll
            for (int v = 0; v < 8; ++v) {
                float x = acc[mi][s][v] + bv[v];
                x = x > 0.0f ? x : 0.0f;
                o.v[v] = (__bf16)x;
            }
            long pix = ntile * 64 + s * 16 + nl;
            *(uint4*)(out + (size_t)pix * Cout + mt * 16 + half * 8) = o.q;
        }
    }
}

// =====================================================================================
// Direct (no-LDS) WMMA conv for the tiny Cout=1 heads (padded to 16 couts).
// ACT: 1 = sigmoid -> fp32 scalar (cout 0), 2 = tanh -> fp32 scalar (cout 0)
// =====================================================================================
template<int ACT>
__global__ __launch_bounds__(256) void conv3x3_wmma_head(
    const __bf16* __restrict__ in,
    const __bf16* __restrict__ wp,
    const float*  __restrict__ bias,
    float*        __restrict__ outF,
    int P, int Cin, long totalWaves)
{
    const int lane = threadIdx.x & 31;
    const int wid  = __builtin_amdgcn_readfirstlane(threadIdx.x >> 5);
    const long waveId = (long)blockIdx.x * 8 + wid;
    if (waveId >= totalWaves) return;

    const long ntile = waveId;            // Mtiles == 1
    const int  half  = lane >> 4;
    const int  nl    = lane & 15;
    const int  CinBlk = Cin >> 5;
    const int  PP = P * P;

    int pg[4], py[4], px[4];
#pragma unroll
    for (int s = 0; s < 4; ++s) {
        long pix = ntile * 64 + s * 16 + nl;
        pg[s] = (int)(pix / PP);
        int w = (int)(pix % PP);
        py[s] = w / P;
        px[s] = w % P;
    }

    v8f acc[4] = {};
    const __bf16* wbase = wp + lane * 16;

    for (int kk = 0; kk < 9; ++kk) {
        const int dy = kk / 3 - 1;
        const int dx = kk % 3 - 1;
        for (int cb = 0; cb < CinBlk; ++cb) {
            BF16x16 a;
            const uint4* ap = (const uint4*)(wbase + (size_t)(kk * CinBlk + cb) * 512);
            a.q[0] = ap[0];
            a.q[1] = ap[1];

            BF16x16 b[4];
#pragma unroll
            for (int s = 0; s < 4; ++s) {
                uint4 lo = {0, 0, 0, 0};
                uint4 hi = {0, 0, 0, 0};
                int yy = py[s] + dy;
                int xx = px[s] + dx;
                bool valid = (yy >= 0) && (yy < P) && (xx >= 0) && (xx < P);
                if (valid) {
                    size_t off = ((size_t)pg[s] * PP + (size_t)yy * P + xx) * (size_t)Cin;
                    const __bf16* bp = in + off + cb * 32 + half * 8;
                    lo = *(const uint4*)bp;
                    hi = *(const uint4*)(bp + 16);
                }
                b[s].q[0] = lo;
                b[s].q[1] = hi;
            }
#pragma unroll
            for (int s = 0; s < 4; ++s) {
                acc[s] = __builtin_amdgcn_wmma_f32_16x16x32_bf16(
                    false, a.v, false, b[s].v, (short)0, acc[s], false, false);
            }
        }
    }

    float b0 = bias[0];
    if (half == 0) {  // lanes 0..15, VGPR0 holds M==0 (the only real cout)
#pragma unroll
        for (int s = 0; s < 4; ++s) {
            float x = acc[s][0] + b0;
            float r = (ACT == 1) ? (1.0f / (1.0f + __expf(-x))) : tanhf(x);
            long pix = ntile * 64 + s * 16 + nl;
            outF[pix] = r;
        }
    }
}

// ---------------- overlap-add normalization (gather form, <=4 patches/pixel) ----------------
__global__ __launch_bounds__(256) void attention_finalize(
    const float* __restrict__ score, float* __restrict__ att, int total)
{
    int i = blockIdx.x * 256 + threadIdx.x;
    if (i >= total) return;
    int x = i % 192;
    int y = (i / 192) % 192;
    int b = i / (192 * 192);

    int ph_lo = imax(0, (y - 12) / 12);
    int ph_hi = imin(y / 12, 14);
    int pw_lo = imax(0, (x - 12) / 12);
    int pw_hi = imin(x / 12, 14);

    float sum = 0.0f;
    int cnt = 0;
    for (int ph = ph_lo; ph <= ph_hi; ++ph) {
        int yy = y - ph * 12;          // in [0,24)
        for (int pw = pw_lo; pw <= pw_hi; ++pw) {
            int xx = x - pw * 12;      // in [0,24)
            int patch = (b * 15 + ph) * 15 + pw;
            sum += score[(size_t)patch * 576 + yy * 24 + xx];
            ++cnt;
        }
    }
    att[i] = sum / (float)imax(cnt, 1);
}

// ---------------- host launcher ----------------
extern "C" void kernel_launch(void* const* d_in, const int* in_sizes, int n_in,
                              void* d_out, int out_size, void* d_ws, size_t ws_size,
                              hipStream_t stream)
{
    (void)in_sizes; (void)n_in; (void)out_size; (void)ws_size;

    const float* src = (const float*)d_in[0];
    const float* tgt = (const float*)d_in[1];
    const float* W1  = (const float*)d_in[2];  const float* b1  = (const float*)d_in[3];
    const float* W2  = (const float*)d_in[4];  const float* b2  = (const float*)d_in[5];
    const float* W3  = (const float*)d_in[6];  const float* b3  = (const float*)d_in[7];
    const float* Wr1 = (const float*)d_in[8];  const float* br1 = (const float*)d_in[9];
    const float* Wr2 = (const float*)d_in[10]; const float* br2 = (const float*)d_in[11];
    const float* Wr3 = (const float*)d_in[12]; const float* br3 = (const float*)d_in[13];

    float* outReg = (float*)d_out;             // [2,1,192,192]
    float* outAtt = (float*)d_out + 73728;     // [2,1,192,192]

    char* ws = (char*)d_ws;
    size_t off = 0;
    auto alloc = [&](size_t bytes) {
        size_t o = off;
        off = (off + bytes + 255) & ~(size_t)255;
        return o;
    };

    const size_t NPATCH = 450, PPIX = 576;          // 24*24
    const size_t IMGPIX = (size_t)192 * 192;        // per batch

    size_t oX0  = alloc(2 * IMGPIX * 128 * 2);                 // NHWC bf16 concat input
    size_t oA1  = alloc(NPATCH * PPIX * 512 * 2);              // att act1 (reused as reg act1)
    size_t oA2  = alloc(NPATCH * PPIX * 256 * 2);              // att act2 (reused as reg act2)
    size_t oSC  = alloc(NPATCH * PPIX * 4);                    // sigmoid scores fp32
    size_t oW1  = alloc((size_t)32 * 36  * 512 * 2);           // 512x(128*9)
    size_t oW2  = alloc((size_t)16 * 144 * 512 * 2);           // 256x(512*9)
    size_t oW3  = alloc((size_t)1  * 72  * 512 * 2);           // 16(pad)x(256*9)
    size_t oWr1 = alloc((size_t)32 * 36  * 512 * 2);           // 512x(128*9)
    size_t oWr2 = alloc((size_t)4  * 144 * 512 * 2);           // 64x(512*9)
    size_t oWr3 = alloc((size_t)1  * 18  * 512 * 2);           // 16(pad)x(64*9)

    __bf16* X0  = (__bf16*)(ws + oX0);
    __bf16* A1  = (__bf16*)(ws + oA1);
    __bf16* A2  = (__bf16*)(ws + oA2);
    float*  SC  = (float*)(ws + oSC);
    __bf16* W1p = (__bf16*)(ws + oW1);
    __bf16* W2p = (__bf16*)(ws + oW2);
    __bf16* W3p = (__bf16*)(ws + oW3);
    __bf16* Wr1p= (__bf16*)(ws + oWr1);
    __bf16* Wr2p= (__bf16*)(ws + oWr2);
    __bf16* Wr3p= (__bf16*)(ws + oWr3);

    // ---- packing ----
    {
        int total = 2 * 192 * 192 * 128;
        pack_input_nhwc<<<(total + 255) / 256, 256, 0, stream>>>(src, tgt, X0, total);
    }
    auto packW = [&](const float* W, __bf16* out, int Cout, int Cin, int Mtiles) {
        int total = Mtiles * 9 * (Cin >> 5) * 512;
        pack_weights_frag<<<(total + 255) / 256, 256, 0, stream>>>(W, out, Cout, Cin, total);
    };
    packW(W1,  W1p,  512, 128, 32);
    packW(W2,  W2p,  256, 512, 16);
    packW(W3,  W3p,    1, 256,  1);
    packW(Wr1, Wr1p, 512, 128, 32);
    packW(Wr2, Wr2p,  64, 512,  4);
    packW(Wr3, Wr3p,   1,  64,  1);

    // ---- attention branch ----
    {   // conv1: patch-gather 128 -> 512, P=24, N = 450*576 = 259200, ntiles=4050
        int ntiles = 259200 / 64, MB = 2;  // Mtiles=32 -> 2 groups of 16
        conv3x3_wmma_lds<1><<<ntiles * MB, 256, 0, stream>>>(
            X0, W1p, b1, A1, 24, 128, 512, 32, MB);
    }
    {   // conv2: 512 -> 256
        int ntiles = 259200 / 64, MB = 1;  // Mtiles=16
        conv3x3_wmma_lds<0><<<ntiles * MB, 256, 0, stream>>>(
            A1, W2p, b2, A2, 24, 512, 256, 16, MB);
    }
    {   // conv3: 256 -> 1 (padded to 16), sigmoid -> fp32 score
        long waves = 259200L / 64;
        conv3x3_wmma_head<1><<<(int)((waves + 7) / 8), 256, 0, stream>>>(
            A2, W3p, b3, SC, 24, 256, waves);
    }
    {   // overlap-add + count normalization
        int total = 2 * 192 * 192;
        attention_finalize<<<(total + 255) / 256, 256, 0, stream>>>(SC, outAtt, total);
    }

    // ---- registration branch (reuses A1/A2 regions) ----
    {   // conv1: 128 -> 512 on full 192x192, N = 2*36864 = 73728, ntiles=1152
        int ntiles = 73728 / 64, MB = 2;   // Mtiles=32
        conv3x3_wmma_lds<0><<<ntiles * MB, 256, 0, stream>>>(
            X0, Wr1p, br1, A1, 192, 128, 512, 32, MB);
    }
    {   // conv2: 512 -> 64
        int ntiles = 73728 / 64, MB = 1;   // Mtiles=4 (waves 2..7 stage only)
        conv3x3_wmma_lds<0><<<ntiles * MB, 256, 0, stream>>>(
            A1, Wr2p, br2, A2, 192, 512, 64, 4, MB);
    }
    {   // conv3: 64 -> 1 (padded to 16), tanh -> fp32 registered
        long waves = 73728L / 64;
        conv3x3_wmma_head<2><<<(int)((waves + 7) / 8), 256, 0, stream>>>(
            A2, Wr3p, br3, outReg, 192, 64, waves);
    }
}
